// AttentionNTKChoiceModel_79620103733795
// MI455X (gfx1250) — matrix operand order repaired and checked by reading, not compiled, not run on hardware
//
#include <hip/hip_runtime.h>
#include <hip/hip_bf16.h>

// ---------------------------------------------------------------------------
// AttentionNTKChoiceModel for MI455X (gfx1250, wave32, WMMA bf16 path)
// ---------------------------------------------------------------------------

typedef __attribute__((ext_vector_type(16))) __bf16        v16bf;
typedef __attribute__((ext_vector_type(8)))  float         v8f;
typedef __attribute__((ext_vector_type(4)))  unsigned int  u32x4;
typedef __attribute__((ext_vector_type(4)))  float         f32x4;

#define B_   32
#define L_   512
#define D_   128
#define H_   8
#define D0_  32
#define NEGV    (-10000.0f)
#define SCALE1  (0.17677669529663687f)  // 1/sqrt(32)
#define SCALEH  (0.35355339059327373f)  // 1/sqrt(8)

// Workspace layout (bytes)
static constexpr size_t XBF_OFF  = 0;                                    // X bf16: 16384x128
static constexpr size_t W1BF_OFF = XBF_OFF  + (size_t)B_*L_*D_*2;        // 4 MB
static constexpr size_t W2BF_OFF = W1BF_OFF + (size_t)256*D_*2;
static constexpr size_t W3BF_OFF = W2BF_OFF + (size_t)256*D_*2;
static constexpr size_t QBF_OFF  = W3BF_OFF + (size_t)256*D_*2;
static constexpr size_t KBF_OFF  = QBF_OFF  + (size_t)B_*H_*L_*D0_*2;    // 8 MB each
static constexpr size_t VT_OFF   = KBF_OFF  + (size_t)B_*H_*L_*D0_*2;
static constexpr size_t VBUF_OFF = VT_OFF   + (size_t)B_*H_*L_*D0_*2;    // v[B,L] f32

union Frag16 { u32x4 u[2]; v16bf v; __bf16 h[16]; };

__device__ inline v8f wmma_bf16(v16bf a, v16bf b, v8f c) {
  // (neg_a, A, neg_b, B, c_mod, C, reuse_a, reuse_b)
  return __builtin_amdgcn_wmma_f32_16x16x32_bf16(false, a, false, b, (short)0, c,
                                                 false, false);
}

// ---------------------------------------------------------------------------
// Kernel 0: fp32 -> bf16 conversion of X and W1/W2/W3; zero v-accumulator.
// ---------------------------------------------------------------------------
__global__ void prep_kernel(const float* __restrict__ X,
                            const float* __restrict__ W1,
                            const float* __restrict__ W2,
                            const float* __restrict__ W3,
                            __bf16* __restrict__ Xbf,
                            __bf16* __restrict__ W1bf,
                            __bf16* __restrict__ W2bf,
                            __bf16* __restrict__ W3bf,
                            float* __restrict__ vbuf) {
  int i = blockIdx.x * blockDim.x + threadIdx.x;
  const int NX = B_ * L_ * D_;   // 2097152
  const int NW = 256 * D_;       // 32768
  if (i < NX) Xbf[i] = (__bf16)X[i];
  if (i < NW) {
    W1bf[i] = (__bf16)W1[i];
    W2bf[i] = (__bf16)W2[i];
    W3bf[i] = (__bf16)W3[i];
  }
  if (i < B_ * L_) vbuf[i] = 0.0f;
}

// ---------------------------------------------------------------------------
// Kernel 1: projections Q = X W1^T, K = X W2^T, V = X W3^T  (bf16 WMMA)
// One wave computes a 16x64 strip (4 col tiles): A-fragments (X) loaded once
// and reused 4x, 16 WMMAs per wave.  Q,K stored [B][H][L][32]; V transposed
// [B][H][32][L] so attention B-fragments are contiguous 32-byte loads.
// ---------------------------------------------------------------------------
__global__ void proj_kernel(const __bf16* __restrict__ Xbf,
                            const __bf16* __restrict__ W1bf,
                            const __bf16* __restrict__ W2bf,
                            const __bf16* __restrict__ W3bf,
                            __bf16* __restrict__ Qbf,
                            __bf16* __restrict__ Kbf,
                            __bf16* __restrict__ Vtbf) {
  const int lane = threadIdx.x;
  const int m  = lane & 15;
  const int hl = lane >> 4;
  const int rt  = blockIdx.x;   // 0..1023 row tiles of 16 over B*L
  const int ctg = blockIdx.y;   // 0..3   groups of 4 col tiles (64 cols)
  const int mat = blockIdx.z;   // 0:Q 1:K 2:V
  const __bf16* W = (mat == 0) ? W1bf : (mat == 1) ? W2bf : W3bf;
  __bf16* Out = (mat == 0) ? Qbf : (mat == 1) ? Kbf : Vtbf;
  const bool transposed = (mat == 2);

  // A fragments for all four K-steps (D = 128 = 4 * 32), loaded once.
  const __bf16* Xrow = Xbf + (size_t)(rt * 16 + m) * D_;
  Frag16 a[4];
#pragma unroll
  for (int kt = 0; kt < 4; ++kt) {
    a[kt].u[0] = *(const u32x4*)(Xrow + kt * 32 + hl * 16);
    a[kt].u[1] = *(const u32x4*)(Xrow + kt * 32 + hl * 16 + 8);
  }

#pragma unroll
  for (int ct = 0; ct < 4; ++ct) {
    const int col = (ctg * 4 + ct) * 16 + m;          // global col in [0,256)
    const __bf16* Wrow = W + (size_t)col * D_;
    v8f c = {};
#pragma unroll
    for (int kt = 0; kt < 4; ++kt) {
      Frag16 bw;  // B frag: column n = m, contiguous K [kt*32+hl*16 .. +15]
      bw.u[0] = *(const u32x4*)(Wrow + kt * 32 + hl * 16);
      bw.u[1] = *(const u32x4*)(Wrow + kt * 32 + hl * 16 + 8);
      c = wmma_bf16(a[kt].v, bw.v, c);
    }
    const int hh = col >> 5, dd = col & 31;
#pragma unroll
    for (int j = 0; j < 8; ++j) {
      int grow = rt * 16 + j + 8 * hl;                // global row in [0,B*L)
      int bb = grow >> 9, ll = grow & 511;
      size_t bh = (size_t)(bb * H_ + hh);
      size_t idx = transposed ? (bh * D0_ + dd) * L_ + ll
                              : (bh * L_ + ll) * D0_ + dd;
      Out[idx] = (__bf16)c[j];
    }
  }
}

// ---------------------------------------------------------------------------
// Kernel 2: attention per (b, h, 16-query tile), one wave per block.
//  A: 32x WMMA -> full 16x512 masked/scaled score strip in LDS (f32)
//  B: row softmax (wave32 shfl reductions); exp() written back IN PLACE as
//     bf16 over the same LDS rows (safe: stores depend on the row max, which
//     depends on all loads) -> 32 KB LDS total, ~10 blocks resident per WGP.
//  C: P @ V^T via 2x16 WMMA, weight-dot, atomicAdd into v[B,L]
// ---------------------------------------------------------------------------
__global__ void attn_kernel(const __bf16* __restrict__ Qbf,
                            const __bf16* __restrict__ Kbf,
                            const __bf16* __restrict__ Vtbf,
                            const unsigned char* __restrict__ mask,  // jax bool = 1 byte
                            const float* __restrict__ wvec,          // [H][32]
                            float* __restrict__ vbuf) {              // [B][L]
  __shared__ __attribute__((aligned(16))) float sc[16][L_];   // 32 KB
  __shared__ float rsum[16];

  const int lane = threadIdx.x;
  const int m  = lane & 15;
  const int hl = lane >> 4;
  const int qt = blockIdx.x;   // 0..31
  const int h  = blockIdx.y;   // 0..7
  const int b  = blockIdx.z;   // 0..31
  const int bh = b * H_ + h;

  // Q A-fragment (rows qt*16 .. qt*16+15, K = d0 = 32), loaded once.
  const __bf16* Qrow = Qbf + ((size_t)bh * L_ + qt * 16 + m) * D0_;
  Frag16 aq;
  aq.u[0] = *(const u32x4*)(Qrow + hl * 16);
  aq.u[1] = *(const u32x4*)(Qrow + hl * 16 + 8);

  unsigned rmbits = 0;
#pragma unroll
  for (int j = 0; j < 8; ++j)
    rmbits |= (mask[b * L_ + qt * 16 + j + 8 * hl] ? 1u : 0u) << j;

  // ---- Phase A: scores ----
  const __bf16* Kbase = Kbf + (size_t)bh * L_ * D0_ + (size_t)m * D0_ + hl * 16;
  for (int kt = 0; kt < 32; ++kt) {
    const __bf16* Krow = Kbase + (size_t)kt * 16 * D0_;
    if (kt + 1 < 32)  // gfx1250 global_prefetch of the next K tile
      __builtin_prefetch(Krow + 16 * D0_, 0, 3);
    Frag16 bk;
    bk.u[0] = ((const u32x4*)Krow)[0];
    bk.u[1] = ((const u32x4*)Krow)[1];
    v8f s = {};
    s = wmma_bf16(aq.v, bk.v, s);
    bool cm = mask[b * L_ + kt * 16 + m] != 0;
#pragma unroll
    for (int j = 0; j < 8; ++j) {
      float val = (cm && ((rmbits >> j) & 1u)) ? s[j] * SCALE1 : NEGV;
      sc[j + 8 * hl][kt * 16 + m] = val;
    }
  }
  __syncthreads();

  // ---- Phase B: row softmax, exp stored in place as bf16 (P matrix) ----
  for (int r = 0; r < 16; ++r) {
    f32x4 x0 = *(const f32x4*)&sc[r][lane * 16 + 0];
    f32x4 x1 = *(const f32x4*)&sc[r][lane * 16 + 4];
    f32x4 x2 = *(const f32x4*)&sc[r][lane * 16 + 8];
    f32x4 x3 = *(const f32x4*)&sc[r][lane * 16 + 12];
    float xs[16];
#pragma unroll
    for (int i = 0; i < 4; ++i) {
      xs[i] = x0[i]; xs[4 + i] = x1[i]; xs[8 + i] = x2[i]; xs[12 + i] = x3[i];
    }
    float mx = NEGV;
#pragma unroll
    for (int i = 0; i < 16; ++i) mx = fmaxf(mx, xs[i]);
#pragma unroll
    for (int o = 16; o >= 1; o >>= 1) mx = fmaxf(mx, __shfl_xor(mx, o, 32));
    float sum = 0.0f;
    Frag16 pw;
#pragma unroll
    for (int i = 0; i < 16; ++i) {
      float e = __expf(xs[i] - mx);
      pw.h[i] = (__bf16)e;
      sum += e;
    }
    __bf16* prow = (__bf16*)&sc[r][0];            // in-place bf16 overlay
    ((u32x4*)(prow + lane * 16))[0] = pw.u[0];
    ((u32x4*)(prow + lane * 16))[1] = pw.u[1];
#pragma unroll
    for (int o = 16; o >= 1; o >>= 1) sum += __shfl_xor(sum, o, 32);
    if (lane == 0) rsum[r] = sum;
  }
  __syncthreads();

  // ---- Phase C: O = P @ V^T, then dot with w[h,:] ----
  float lanepart[8];
#pragma unroll
  for (int j = 0; j < 8; ++j) lanepart[j] = 0.0f;

  const __bf16* prowm = (const __bf16*)&sc[m][0];  // P row m (bf16 overlay)
  for (int dt = 0; dt < 2; ++dt) {
    v8f acc = {};
    for (int kt = 0; kt < 16; ++kt) {
      Frag16 ap, bv;
      ap.u[0] = *(const u32x4*)(prowm + kt * 32 + hl * 16);
      ap.u[1] = *(const u32x4*)(prowm + kt * 32 + hl * 16 + 8);
      const __bf16* Vrow =
          Vtbf + ((size_t)bh * D0_ + dt * 16 + m) * L_ + kt * 32 + hl * 16;
      bv.u[0] = ((const u32x4*)Vrow)[0];
      bv.u[1] = ((const u32x4*)Vrow)[1];
      acc = wmma_bf16(ap.v, bv.v, acc);
    }
    float wv = wvec[h * D0_ + dt * 16 + m] * SCALEH;
#pragma unroll
    for (int j = 0; j < 8; ++j)
      lanepart[j] += (acc[j] / rsum[j + 8 * hl]) * wv;
  }

  // reduce across the 16 lanes of each half (cols of the tile)
#pragma unroll
  for (int j = 0; j < 8; ++j) {
#pragma unroll
    for (int o = 8; o >= 1; o >>= 1)
      lanepart[j] += __shfl_xor(lanepart[j], o, 32);
  }
  if (m == 0) {
#pragma unroll
    for (int j = 0; j < 8; ++j)
      atomicAdd(&vbuf[b * L_ + qt * 16 + j + 8 * hl], lanepart[j]);
  }
}

// ---------------------------------------------------------------------------
// Kernel 3: masked softmax over L per batch -> probs (fp32 output)
// ---------------------------------------------------------------------------
__global__ void final_softmax_kernel(const float* __restrict__ vbuf,
                                     const unsigned char* __restrict__ mask,
                                     float* __restrict__ out) {
  __shared__ float red[L_];
  const int b = blockIdx.x;
  const int t = threadIdx.x;
  float val = mask[b * L_ + t] ? vbuf[b * L_ + t] : NEGV;
  red[t] = val;
  __syncthreads();
  for (int s = 256; s >= 1; s >>= 1) {
    if (t < s) red[t] = fmaxf(red[t], red[t + s]);
    __syncthreads();
  }
  float mx = red[0];
  __syncthreads();
  float e = __expf(val - mx);
  red[t] = e;
  __syncthreads();
  for (int s = 256; s >= 1; s >>= 1) {
    if (t < s) red[t] += red[t + s];
    __syncthreads();
  }
  out[b * L_ + t] = e / red[0];
}

// ---------------------------------------------------------------------------
extern "C" void kernel_launch(void* const* d_in, const int* in_sizes, int n_in,
                              void* d_out, int out_size, void* d_ws, size_t ws_size,
                              hipStream_t stream) {
  const float*         X    = (const float*)d_in[0];
  const unsigned char* mask = (const unsigned char*)d_in[1];  // jax bool: 1 byte
  const float*         W1   = (const float*)d_in[2];
  const float*         W2   = (const float*)d_in[3];
  const float*         W3   = (const float*)d_in[4];
  const float*         wv   = (const float*)d_in[5];

  char* ws = (char*)d_ws;
  __bf16* Xbf  = (__bf16*)(ws + XBF_OFF);
  __bf16* W1bf = (__bf16*)(ws + W1BF_OFF);
  __bf16* W2bf = (__bf16*)(ws + W2BF_OFF);
  __bf16* W3bf = (__bf16*)(ws + W3BF_OFF);
  __bf16* Qbf  = (__bf16*)(ws + QBF_OFF);
  __bf16* Kbf  = (__bf16*)(ws + KBF_OFF);
  __bf16* Vtbf = (__bf16*)(ws + VT_OFF);
  float*  vbuf = (float*)(ws + VBUF_OFF);

  // 1) convert + zero
  prep_kernel<<<(B_ * L_ * D_ + 255) / 256, 256, 0, stream>>>(
      X, W1, W2, W3, Xbf, W1bf, W2bf, W3bf, vbuf);

  // 2) QKV projections: 1024 row tiles x 4 col groups x 3 matrices, 1 wave each
  proj_kernel<<<dim3(1024, 4, 3), 32, 0, stream>>>(
      Xbf, W1bf, W2bf, W3bf, Qbf, Kbf, Vtbf);

  // 3) attention: (qtile, head, batch), 1 wave each
  attn_kernel<<<dim3(L_ / 16, H_, B_), 32, 0, stream>>>(
      Qbf, Kbf, Vtbf, mask, wv, vbuf);

  // 4) final masked softmax over L
  final_softmax_kernel<<<B_, L_, 0, stream>>>(vbuf, mask, (float*)d_out);
}
